// GNN_inGeo_14946486190735
// MI455X (gfx1250) — matrix operand order repaired and checked by reading, not compiled, not run on hardware
//
#include <hip/hip_runtime.h>

typedef __attribute__((ext_vector_type(16))) _Float16 v16h;
typedef __attribute__((ext_vector_type(8)))  _Float16 v8h;
typedef __attribute__((ext_vector_type(4)))  _Float16 v4h;
typedef __attribute__((ext_vector_type(8)))  float    v8f;
typedef __attribute__((ext_vector_type(4)))  float    v4f;

#define NNODES   4096
#define NBATCH   64
#define FEAT     64
#define TILE_OUT 112          // output rows per block
#define ROWS     128          // extended rows (TILE_OUT + 16-row halo)
#define STRIDE   72           // padded f16 row stride (36 dwords -> no bank conflicts)

// f16-element offsets inside dynamic LDS
#define WT_OFF   0                              // 6 * 64 * STRIDE  (transposed weights)
#define XS_OFF   (WT_OFF + 6*64*STRIDE)         // x tile (f16)
#define P_OFF    (XS_OFF + ROWS*STRIDE)         // pooled activations (p1, then p2)
#define H1_OFF   (P_OFF  + ROWS*STRIDE)         // h1 tile (f16)
#define F16_TOT  (H1_OFF + ROWS*STRIDE)
#define BIAS_OFF_BYTES (F16_TOT * 2)            // 4 * 64 f32 biases
#define SMEM_BYTES (BIAS_OFF_BYTES + 4*64*4)    // = 111616 bytes

// pack f16 runs p[0..7] and p[16..23] from LDS into a v16h (two ds_load_b128)
__device__ __forceinline__ v16h pack16(const _Float16* p) {
  v8h u0 = *(const v8h*)p;
  v8h u1 = *(const v8h*)(p + 16);
  v16h r;
#pragma unroll
  for (int i = 0; i < 8; ++i) { r[i] = u0[i]; r[8 + i] = u1[i]; }
  return r;
}

// A-fragment (16x32 f16, chunk c of K): lane holds row M=lane&15.
// VGPR0-3: K = 32c + 8*hi + {0..7}; VGPR4-7: K = 32c + 16 + 8*hi + {0..7}
__device__ __forceinline__ v16h load_afrag(const _Float16* rowptr, int c, int hi) {
  const _Float16* p = rowptr + 32 * c + (hi << 3);
  v8h u0 = *(const v8h*)p;
  v8h u1 = *(const v8h*)(p + 16);
  v16h r;
#pragma unroll
  for (int i = 0; i < 8; ++i) { r[i] = u0[i]; r[8 + i] = u1[i]; }
  return r;
}

// One 16x16 C-tile of A(16x64) @ W(64x64): two K=32 WMMAs.
// Weights stored transposed in LDS: wbase[n*STRIDE + k] = W[k][n].
// B-fragment: lane col n = (lane&15)+16*nt, K = 32c + 8*hi + {0..7}, then 32c + 16 + 8*hi + {0..7}.
__device__ __forceinline__ v8f tileMM(v16h a0, v16h a1, const _Float16* wbase,
                                      int col, int hi, v8f acc) {
  const _Float16* p = wbase + col * STRIDE + (hi << 3);
  v16h b0 = pack16(p);
  v16h b1 = pack16(p + 32);
  acc = __builtin_amdgcn_wmma_f32_16x16x32_f16(false, a0, false, b0, (short)0, acc, false, false);
  acc = __builtin_amdgcn_wmma_f32_16x16x32_f16(false, a1, false, b1, (short)0, acc, false, false);
  return acc;
}

__global__ __launch_bounds__(256)
void sage2_fused_kernel(const float* __restrict__ loc,
                        const float* __restrict__ Wp1, const float* __restrict__ bp1,
                        const float* __restrict__ Ws1, const float* __restrict__ Wn1,
                        const float* __restrict__ b1,
                        const float* __restrict__ Wp2, const float* __restrict__ bp2,
                        const float* __restrict__ Ws2, const float* __restrict__ Wn2,
                        const float* __restrict__ b2,
                        float* __restrict__ out) {
  extern __shared__ __align__(16) char smem_raw[];
  _Float16* sh   = (_Float16*)smem_raw;
  float*    sbia = (float*)(smem_raw + BIAS_OFF_BYTES);

  const int tid = threadIdx.x;
  const int b   = blockIdx.y;
  const int n0  = blockIdx.x * TILE_OUT;

  // ---------------- stage 0: cooperative LDS fill ----------------
  {
    const float* Wsrc[6] = {Wp1, Ws1, Wn1, Wp2, Ws2, Wn2};
#pragma unroll
    for (int m = 0; m < 6; ++m) {
      const float* W = Wsrc[m];
      _Float16* dst = sh + WT_OFF + m * 64 * STRIDE;
      for (int i = tid; i < 64 * 64; i += 256) {
        int k = i >> 6, n = i & 63;
        dst[n * STRIDE + k] = (_Float16)W[i];       // transpose + f32->f16
      }
    }
    const float* Bsrc[4] = {bp1, b1, bp2, b2};
    {
      int i = tid;                                   // 256 threads, 256 values
      sbia[i] = Bsrc[i >> 6][i & 63];
    }
    // x tile: ROWS x 64 f32, row clamped into [0, N-1]
    for (int t4 = tid; t4 < ROWS * 16; t4 += 256) {
      int row = t4 >> 4;
      int c4  = (t4 & 15) << 2;
      int g = n0 - 16 + row;
      g = g < 0 ? 0 : (g >= NNODES ? NNODES - 1 : g);
      v4f xv = *(const v4f*)(loc + ((size_t)b * NNODES + g) * FEAT + c4);
      v4h hv;
      hv.x = (_Float16)xv.x; hv.y = (_Float16)xv.y;
      hv.z = (_Float16)xv.z; hv.w = (_Float16)xv.w;
      *(v4h*)(sh + XS_OFF + row * STRIDE + c4) = hv;
    }
  }
  __syncthreads();

  const int lane = tid & 31;
  const int wv   = tid >> 5;        // 8 waves, strip = 16 rows
  const int mrow = lane & 15;
  const int hi   = lane >> 4;
  const int Rb   = wv << 4;         // strip base (local row)
  const v16h zfrag = {};

  const _Float16* wWp1 = sh + WT_OFF + 0 * 64 * STRIDE;
  const _Float16* wWs1 = sh + WT_OFF + 1 * 64 * STRIDE;
  const _Float16* wWn1 = sh + WT_OFF + 2 * 64 * STRIDE;
  const _Float16* wWp2 = sh + WT_OFF + 3 * 64 * STRIDE;
  const _Float16* wWs2 = sh + WT_OFF + 4 * 64 * STRIDE;
  const _Float16* wWn2 = sh + WT_OFF + 5 * 64 * STRIDE;

  // is this lane's A-row the zero-in-degree node (global row 0)?
  const bool zrow = (n0 - 16 + Rb + mrow) == 0;

  // ---------------- stage 1a: p1 = relu(x @ Wp1 + bp1) ----------------
  const _Float16* xsrow = sh + XS_OFF + (Rb + mrow) * STRIDE;
  v16h xa0 = load_afrag(xsrow, 0, hi);
  v16h xa1 = load_afrag(xsrow, 1, hi);
#pragma unroll
  for (int nt = 0; nt < 4; ++nt) {
    int col = mrow + (nt << 4);
    v8f acc = {};
    acc = tileMM(xa0, xa1, wWp1, col, hi, acc);
    float bias = sbia[0 * 64 + col];
#pragma unroll
    for (int v = 0; v < 8; ++v) {
      float val = acc[v] + bias;
      val = val > 0.f ? val : 0.f;
      sh[P_OFF + (Rb + v + (hi << 3)) * STRIDE + col] = (_Float16)val;
    }
  }
  __syncthreads();

  // ---------------- stage 1b: h1 = x@Ws1 + shift(p1)@Wn1 + b1 ----------------
  {
    int pr = Rb + mrow - 1; if (pr < 0) pr = 0;     // row -1 only feeds unused halo
    v16h pa0 = load_afrag(sh + P_OFF + pr * STRIDE, 0, hi);
    v16h pa1 = load_afrag(sh + P_OFF + pr * STRIDE, 1, hi);
    if (zrow) { pa0 = zfrag; pa1 = zfrag; }         // zero-in-degree node
#pragma unroll
    for (int nt = 0; nt < 4; ++nt) {
      int col = mrow + (nt << 4);
      v8f acc = {};
      acc = tileMM(xa0, xa1, wWs1, col, hi, acc);
      acc = tileMM(pa0, pa1, wWn1, col, hi, acc);
      float bias = sbia[1 * 64 + col];
#pragma unroll
      for (int v = 0; v < 8; ++v) {
        float val = acc[v] + bias;
        sh[H1_OFF + (Rb + v + (hi << 3)) * STRIDE + col] = (_Float16)val;
      }
    }
  }
  __syncthreads();

  // ---------------- stage 2a: p2 = relu(h1 @ Wp2 + bp2)  (reuses P buffer) ----------------
  const _Float16* h1row = sh + H1_OFF + (Rb + mrow) * STRIDE;
  v16h ha0 = load_afrag(h1row, 0, hi);
  v16h ha1 = load_afrag(h1row, 1, hi);
#pragma unroll
  for (int nt = 0; nt < 4; ++nt) {
    int col = mrow + (nt << 4);
    v8f acc = {};
    acc = tileMM(ha0, ha1, wWp2, col, hi, acc);
    float bias = sbia[2 * 64 + col];
#pragma unroll
    for (int v = 0; v < 8; ++v) {
      float val = acc[v] + bias;
      val = val > 0.f ? val : 0.f;
      sh[P_OFF + (Rb + v + (hi << 3)) * STRIDE + col] = (_Float16)val;
    }
  }
  __syncthreads();

  // ---------------- stage 2b: h2 = h1@Ws2 + shift(p2)@Wn2 + b2 -> global ----------------
  if (wv >= 1) {                                    // local rows 16..127 = outputs n0..n0+111
    int pr = Rb + mrow - 1;                         // >= 15 here
    v16h qa0 = load_afrag(sh + P_OFF + pr * STRIDE, 0, hi);
    v16h qa1 = load_afrag(sh + P_OFF + pr * STRIDE, 1, hi);
    if (zrow) { qa0 = zfrag; qa1 = zfrag; }
#pragma unroll
    for (int nt = 0; nt < 4; ++nt) {
      int col = mrow + (nt << 4);
      v8f acc = {};
      acc = tileMM(ha0, ha1, wWs2, col, hi, acc);
      acc = tileMM(qa0, qa1, wWn2, col, hi, acc);
      float bias = sbia[3 * 64 + col];
#pragma unroll
      for (int v = 0; v < 8; ++v) {
        int g = n0 - 16 + Rb + v + (hi << 3);
        if (g < NNODES) {
          out[((size_t)b * NNODES + g) * FEAT + col] = acc[v] + bias;
        }
      }
    }
  }
}

extern "C" void kernel_launch(void* const* d_in, const int* in_sizes, int n_in,
                              void* d_out, int out_size, void* d_ws, size_t ws_size,
                              hipStream_t stream) {
  (void)in_sizes; (void)n_in; (void)out_size; (void)d_ws; (void)ws_size;
  const float* loc = (const float*)d_in[0];
  const float* Wp1 = (const float*)d_in[1];
  const float* bp1 = (const float*)d_in[2];
  const float* Ws1 = (const float*)d_in[3];
  const float* Wn1 = (const float*)d_in[4];
  const float* b1  = (const float*)d_in[5];
  const float* Wp2 = (const float*)d_in[6];
  const float* bp2 = (const float*)d_in[7];
  const float* Ws2 = (const float*)d_in[8];
  const float* Wn2 = (const float*)d_in[9];
  const float* b2  = (const float*)d_in[10];
  float* out = (float*)d_out;

  // allow >64KB dynamic LDS (gfx1250 WGP has 320KB)
  (void)hipFuncSetAttribute((const void*)sage2_fused_kernel,
                            hipFuncAttributeMaxDynamicSharedMemorySize, SMEM_BYTES);

  dim3 grid((NNODES + TILE_OUT - 1) / TILE_OUT, NBATCH);
  sage2_fused_kernel<<<grid, dim3(256), SMEM_BYTES, stream>>>(
      loc, Wp1, bp1, Ws1, Wn1, b1, Wp2, bp2, Ws2, Wn2, b2, out);
}
